// MemoryEncoder_86096914415715
// MI455X (gfx1250) — matrix operand order repaired and checked by reading, not compiled, not run on hardware
//
#include <hip/hip_runtime.h>
#include <hip/hip_bf16.h>

// ---------------------------------------------------------------------------
// MemoryEncoder for MI455X (gfx1250, wave32, WMMA).
// All dense math runs through v_wmma_f32_16x16x32_f16 (fp16 in, fp32 acc).
// Cross-lane reductions use DPP16 (no LDS / ds_bpermute stalls).
// ---------------------------------------------------------------------------

typedef _Float16 v16h __attribute__((ext_vector_type(16)));
typedef _Float16 v8h  __attribute__((ext_vector_type(8)));
typedef _Float16 v4h  __attribute__((ext_vector_type(4)));
typedef float    v8f  __attribute__((ext_vector_type(8)));

// Model constants (match reference)
#define C_B    16
#define C_S    1024
#define C_E    2048
#define C_D    512
#define C_NH   8
#define C_LENC 2
#define C_M    16
#define C_LL   24
#define C_KVH  32
#define C_HD   64
#define C_G    4
#define C_KV   (C_KVH * C_HD)   // 2048
#define C_BS   (C_B * C_S)      // 16384

__device__ __forceinline__ v8f wmma_f16(v16h a, v16h b, v8f c) {
  // (neg_a, A, neg_b, B, c_mod, C, reuse_a, reuse_b)
  return __builtin_amdgcn_wmma_f32_16x16x32_f16(false, a, false, b, (short)0, c,
                                                false, false);
}

__device__ __forceinline__ v16h cat8(v8h lo, v8h hi) {
  v16h r;
#pragma unroll
  for (int i = 0; i < 8; i++) { r[i] = lo[i]; r[i + 8] = hi[i]; }
  return r;
}

__device__ __forceinline__ void lds_fence() {
  // CDNA5 split dep counters: wait for all outstanding LDS ops (single-wave WG)
  asm volatile("s_wait_dscnt 0" ::: "memory");
}

// ---- DPP16 cross-lane reductions within each 16-lane row (wave32 halves) ----
// ctrl: 0xB1 quad_perm[1,0,3,2], 0x4E quad_perm[2,3,0,1],
//       0x141 row_half_mirror,   0x140 row_mirror
template <int CTRL>
__device__ __forceinline__ float dpp_f(float x) {
  return __int_as_float(__builtin_amdgcn_update_dpp(
      0, __float_as_int(x), CTRL, 0xF, 0xF, true));
}
__device__ __forceinline__ float rmax16(float x) {
  x = fmaxf(x, dpp_f<0xB1>(x));
  x = fmaxf(x, dpp_f<0x4E>(x));
  x = fmaxf(x, dpp_f<0x141>(x));
  x = fmaxf(x, dpp_f<0x140>(x));
  return x;
}
__device__ __forceinline__ float rsum16(float x) {
  x += dpp_f<0xB1>(x);
  x += dpp_f<0x4E>(x);
  x += dpp_f<0x141>(x);
  x += dpp_f<0x140>(x);
  return x;
}

// ---------------------------------------------------------------------------
// Generic WMMA GEMM:  C[M,N] = act( A[M,K] x W[N,K]^T + bias ) (+ residual)
//   A: fp32 (with optional row-gather) or fp16.  W: fp32 [N,K] row-major.
//   Block: 128 thr = 4 waves. Block tile 64x64, K-step 32.
// ---------------------------------------------------------------------------
template <typename AT, bool GELU, typename OT>
__global__ __launch_bounds__(128) void wmma_gemm_kernel(
    const AT* __restrict__ A, const int* __restrict__ gather,
    const float* __restrict__ W, const float* __restrict__ bias,
    const float* __restrict__ residual, OT* __restrict__ C,
    int Mdim, int N, int K) {
  // 80-byte row stride keeps fragment reads ds_load_b128-aligned.
  __shared__ _Float16 As[64 * 40];
  __shared__ _Float16 Ws[64 * 40];

  const int t    = threadIdx.x;
  const int wave = t >> 5;
  const int lane = t & 31;
  const int hid  = lane >> 4;   // half-wave id (ISA lane-group)
  const int ln16 = lane & 15;
  const long n0 = (long)blockIdx.x * 64;
  const long m0 = (long)blockIdx.y * 64;

  v8f acc[4];
#pragma unroll
  for (int nt = 0; nt < 4; nt++)
#pragma unroll
    for (int v = 0; v < 8; v++) acc[nt][v] = 0.f;

  const bool a_is_f16 = (sizeof(AT) == 2);

  for (int k0 = 0; k0 < K; k0 += 32) {
    // ---- stage A tile (64x32) into LDS as fp16 ----
    if (!a_is_f16) {
#pragma unroll
      for (int i = 0; i < 4; i++) {
        int slot = t + i * 128;               // 512 float4 slots
        int r = slot >> 3, cv = (slot & 7) * 4;
        long row = m0 + r;
        long src = gather ? (long)gather[row] : row;
        const float4 f = *(const float4*)((const float*)A + src * (long)K + k0 + cv);
        v4h hv;
        hv[0] = (_Float16)f.x; hv[1] = (_Float16)f.y;
        hv[2] = (_Float16)f.z; hv[3] = (_Float16)f.w;
        *(v4h*)&As[r * 40 + cv] = hv;
      }
    } else {
#pragma unroll
      for (int i = 0; i < 2; i++) {
        int slot = t + i * 128;               // 256 v8h slots
        int r = slot >> 2, cv = (slot & 3) * 8;
        long row = m0 + r;
        v8h hv = *(const v8h*)((const _Float16*)A + row * (long)K + k0 + cv);
        *(v8h*)&As[r * 40 + cv] = hv;
      }
    }
    // ---- stage W tile (64 cols x 32 k) into LDS as fp16, n-major ----
#pragma unroll
    for (int i = 0; i < 4; i++) {
      int slot = t + i * 128;
      int r = slot >> 3, cv = (slot & 7) * 4;
      const float4 f = *(const float4*)(W + (n0 + r) * (long)K + k0 + cv);
      v4h hv;
      hv[0] = (_Float16)f.x; hv[1] = (_Float16)f.y;
      hv[2] = (_Float16)f.z; hv[3] = (_Float16)f.w;
      *(v4h*)&Ws[r * 40 + cv] = hv;
    }
    if (k0 + 32 < K)  // global_prefetch_b8 of next W k-tile
      __builtin_prefetch((const void*)(W + (n0 + (t >> 1)) * (long)K + k0 + 32), 0, 1);
    __syncthreads();

    // ---- A fragment: row = wave*16 + (lane&15); K halves per ISA layout ----
    const _Float16* ap = &As[(wave * 16 + ln16) * 40];
    v16h af = cat8(*(const v8h*)(ap + hid * 8),
                   *(const v8h*)(ap + 16 + hid * 8));
#pragma unroll
    for (int nt = 0; nt < 4; nt++) {
      const _Float16* bp = &Ws[(nt * 16 + ln16) * 40 + hid * 8];
      v16h bf = cat8(*(const v8h*)bp, *(const v8h*)(bp + 16));
      acc[nt] = wmma_f16(af, bf, acc[nt]);
    }
    __syncthreads();
  }

  // ---- epilogue: bias / GELU / residual, C-layout scatter ----
#pragma unroll
  for (int nt = 0; nt < 4; nt++) {
    long col = n0 + nt * 16 + ln16;
    float bv = bias ? bias[col] : 0.f;
#pragma unroll
    for (int v = 0; v < 8; v++) {
      long row = m0 + wave * 16 + hid * 8 + v;
      float x = acc[nt][v] + bv;
      if (GELU) x = 0.5f * x * (1.f + erff(x * 0.70710678f));
      if (residual) x += residual[row * (long)N + col];
      C[row * (long)N + col] = (OT)x;
    }
  }
}

// ---------------------------------------------------------------------------
// Flash attention, one wave per (q-tile, batch*head). hd = 64, tiles of 16 keys.
//   scores(16x16) = Q(16x64) K^T : two chained WMMAs (K=32 each)
//   online softmax in VALU with DPP16 row reductions (no LDS permutes)
//   P·V via K=16-padded WMMA with V transposed through LDS
// ---------------------------------------------------------------------------
__global__ __launch_bounds__(32) void flash_attn_kernel(
    const _Float16* __restrict__ Q, long qs,
    const _Float16* __restrict__ Kp, const _Float16* __restrict__ Vp, long ks,
    float* __restrict__ O, long os, int nq, int nk, int nheads, float scale) {
  const int bh = blockIdx.y;
  const int b = bh / nheads, h = bh % nheads;
  const _Float16* qb = Q  + (long)b * nq * qs + (long)h * C_HD;
  const _Float16* kb = Kp + (long)b * nk * ks + (long)h * C_HD;
  const _Float16* vb = Vp + (long)b * nk * ks + (long)h * C_HD;
  float* ob = O + (long)b * nq * os + (long)h * C_HD;

  const int lane = threadIdx.x;
  const int hid  = lane >> 4;
  const int ln16 = lane & 15;
  const int q0 = blockIdx.x * 16;

  __shared__ _Float16 Pls[16 * 16];
  __shared__ _Float16 Vt[64 * 16];   // [hd][key]

  // Q fragments (A layout), loaded once
  const _Float16* qrow = qb + (long)(q0 + ln16) * qs;
  v16h qf0 = cat8(*(const v8h*)(qrow + hid * 8),
                  *(const v8h*)(qrow + 16 + hid * 8));
  v16h qf1 = cat8(*(const v8h*)(qrow + 32 + hid * 8),
                  *(const v8h*)(qrow + 48 + hid * 8));

  float mrow[8], lrow[8];
  v8f o[4];
#pragma unroll
  for (int v = 0; v < 8; v++) { mrow[v] = -3.0e38f; lrow[v] = 0.f; }
#pragma unroll
  for (int nt = 0; nt < 4; nt++)
#pragma unroll
    for (int v = 0; v < 8; v++) o[nt][v] = 0.f;

  for (int j = 0; j < nk; j += 16) {
    // K tile as B fragments: lane = key (l&15), hd halves in ISA table order
    const _Float16* krow = kb + (long)(j + ln16) * ks;
    v16h kf0 = cat8(*(const v8h*)(krow + hid * 8),
                    *(const v8h*)(krow + 16 + hid * 8));
    v16h kf1 = cat8(*(const v8h*)(krow + 32 + hid * 8),
                    *(const v8h*)(krow + 48 + hid * 8));
    v8f s;
#pragma unroll
    for (int v = 0; v < 8; v++) s[v] = 0.f;
    s = wmma_f16(qf0, kf0, s);
    s = wmma_f16(qf1, kf1, s);

    float p[8], corr[8];
#pragma unroll
    for (int v = 0; v < 8; v++) {
      float sv = s[v] * scale;
      float mn = fmaxf(mrow[v], rmax16(sv));   // DPP16 row max
      float cc = __expf(mrow[v] - mn);
      float pv = __expf(sv - mn);
      float rs = rsum16(pv);                   // DPP16 row sum
      lrow[v] = lrow[v] * cc + rs;
      mrow[v] = mn;
      p[v] = pv; corr[v] = cc;
    }
#pragma unroll
    for (int nt = 0; nt < 4; nt++)
#pragma unroll
      for (int v = 0; v < 8; v++) o[nt][v] *= corr[v];

    lds_fence();  // previous iteration's Pls/Vt reads complete
    // P: C-layout -> LDS [query][key] (transpose for A fragment)
#pragma unroll
    for (int v = 0; v < 8; v++)
      Pls[(hid * 8 + v) * 16 + ln16] = (_Float16)p[v];
    // V tile transposed into LDS: lane handles key row ln16, hd hid*32..+31
    const _Float16* vrow = vb + (long)(j + ln16) * ks + hid * 32;
#pragma unroll
    for (int c = 0; c < 4; c++) {
      v8h vv = *(const v8h*)(vrow + c * 8);
#pragma unroll
      for (int e = 0; e < 8; e++)
        Vt[(hid * 32 + c * 8 + e) * 16 + ln16] = vv[e];
    }
    lds_fence();

    // P fragment (K = 16 keys, padded to 32)
    v8h plo = *(const v8h*)&Pls[ln16 * 16 + hid * 8];
    v16h pa;
#pragma unroll
    for (int i = 0; i < 8; i++) { pa[i] = plo[i]; pa[i + 8] = (_Float16)0.f; }
#pragma unroll
    for (int nt = 0; nt < 4; nt++) {
      v16h vf;
      {
        v8h vlo = *(const v8h*)&Vt[(nt * 16 + ln16) * 16 + hid * 8];
#pragma unroll
        for (int i = 0; i < 8; i++) { vf[i] = vlo[i]; vf[i + 8] = (_Float16)0.f; }
      }
      o[nt] = wmma_f16(pa, vf, o[nt]);
    }
  }

  float inv[8];
#pragma unroll
  for (int v = 0; v < 8; v++) inv[v] = 1.f / lrow[v];
#pragma unroll
  for (int nt = 0; nt < 4; nt++)
#pragma unroll
    for (int v = 0; v < 8; v++)
      ob[(long)(q0 + hid * 8 + v) * os + nt * 16 + ln16] = o[nt][v] * inv[v];
}

// ---------------------------------------------------------------------------
// LayerNorm over D=512 (one block per row; DPP16 row reduce + LDS combine)
// ---------------------------------------------------------------------------
__global__ __launch_bounds__(128) void layernorm_kernel(
    const float* __restrict__ in, const float* __restrict__ gw,
    const float* __restrict__ bw, float* __restrict__ out) {
  const long row = blockIdx.x;
  const int t = threadIdx.x;
  const float4 v = ((const float4*)(in + row * C_D))[t];
  float s = v.x + v.y + v.z + v.w;
  float q = v.x * v.x + v.y * v.y + v.z * v.z + v.w * v.w;
  s = rsum16(s);
  q = rsum16(q);
  __shared__ float sh[16];
  if ((t & 15) == 0) {       // one lane per 16-lane row: 8 partials
    sh[t >> 4] = s;
    sh[8 + (t >> 4)] = q;
  }
  __syncthreads();
  s = 0.f; q = 0.f;
#pragma unroll
  for (int i = 0; i < 8; i++) { s += sh[i]; q += sh[8 + i]; }
  const float mean = s * (1.f / C_D);
  const float inv = rsqrtf(q * (1.f / C_D) - mean * mean + 1e-5f);
  const float4 g4 = ((const float4*)gw)[t];
  const float4 b4 = ((const float4*)bw)[t];
  float4 o;
  o.x = (v.x - mean) * inv * g4.x + b4.x;
  o.y = (v.y - mean) * inv * g4.y + b4.y;
  o.z = (v.z - mean) * inv * g4.z + b4.z;
  o.w = (v.w - mean) * inv * g4.w + b4.w;
  ((float4*)(out + row * C_D))[t] = o;
}

// Broadcast learned queries (1,M,D) -> (B*M, D)
__global__ __launch_bounds__(128) void bcast_queries_kernel(
    const float* __restrict__ q, float* __restrict__ out) {
  const int r = blockIdx.x, t = threadIdx.x;
  ((float4*)(out + (long)r * C_D))[t] =
      ((const float4*)(q + (long)(r & (C_M - 1)) * C_D))[t];
}

// Gather per-layer group, sigmoid gate, pack fp16: out (2,LL,B,KVH,M,HD)
__global__ __launch_bounds__(256) void gate_out_kernel(
    const _Float16* __restrict__ kg, const _Float16* __restrict__ vg,
    const float* __restrict__ gates, const int* __restrict__ l2g,
    _Float16* __restrict__ out, int total_chunks) {
  int idx = blockIdx.x * 256 + threadIdx.x;
  if (idx >= total_chunks) return;
  int hc = idx & 7;
  int r = idx >> 3;
  int m = r & (C_M - 1);   r >>= 4;
  int kvh = r & (C_KVH-1); r >>= 5;
  int b = r & (C_B - 1);   r >>= 4;
  int ll = r % C_LL;       r /= C_LL;
  int kv = r;  // 0:K 1:V
  int g = l2g[ll];
  const _Float16* src = (kv ? vg : kg) +
      ((long)g * (C_B * C_M) + b * C_M + m) * C_KV + kvh * C_HD + hc * 8;
  float gate = 1.f / (1.f + __expf(-gates[ll * C_KVH + kvh]));
  v8h sv = *(const v8h*)src;
  v8h ov;
#pragma unroll
  for (int e = 0; e < 8; e++) ov[e] = (_Float16)((float)sv[e] * gate);
  *(v8h*)(out + (long)idx * 8) = ov;
}

// ---------------------------------------------------------------------------
// Host orchestration
// ---------------------------------------------------------------------------
extern "C" void kernel_launch(void* const* d_in, const int* in_sizes, int n_in,
                              void* d_out, int out_size, void* d_ws, size_t ws_size,
                              hipStream_t stream) {
  (void)in_sizes; (void)n_in; (void)out_size; (void)ws_size;
  const int*   ids      = (const int*)d_in[0];
  const float* emb      = (const float*)d_in[2];
  const float* ipw      = (const float*)d_in[3];
  const float* ipb      = (const float*)d_in[4];
  const float* enc_in_w = (const float*)d_in[5];
  const float* enc_in_b = (const float*)d_in[6];
  const float* enc_ow   = (const float*)d_in[7];
  const float* enc_ob   = (const float*)d_in[8];
  const float* ln1g     = (const float*)d_in[9];
  const float* ln1b     = (const float*)d_in[10];
  const float* ln2g     = (const float*)d_in[11];
  const float* ln2b     = (const float*)d_in[12];
  const float* ff1w     = (const float*)d_in[13];
  const float* ff1b     = (const float*)d_in[14];
  const float* ff2w     = (const float*)d_in[15];
  const float* ff2b     = (const float*)d_in[16];
  const float* queries  = (const float*)d_in[17];
  const float* rs_in_w  = (const float*)d_in[18];
  const float* rs_in_b  = (const float*)d_in[19];
  const float* rs_ow    = (const float*)d_in[20];
  const float* rs_ob    = (const float*)d_in[21];
  const float* rl1g     = (const float*)d_in[22];
  const float* rl1b     = (const float*)d_in[23];
  const float* rl2g     = (const float*)d_in[24];
  const float* rl2b     = (const float*)d_in[25];
  const float* rf1w     = (const float*)d_in[26];
  const float* rf1b     = (const float*)d_in[27];
  const float* rf2w     = (const float*)d_in[28];
  const float* rf2b     = (const float*)d_in[29];
  const float* pkw      = (const float*)d_in[30];
  const float* pkb      = (const float*)d_in[31];
  const float* pvw      = (const float*)d_in[32];
  const float* pvb      = (const float*)d_in[33];
  const float* gates    = (const float*)d_in[34];
  const int*   l2g      = (const int*)d_in[35];

  // Workspace layout (bytes); needs ~160 MB
  char* ws = (char*)d_ws;
  float*    x       = (float*)(ws);                                // 33.5 MB
  float*    h       = (float*)(ws + (size_t)33554432);             // 33.5 MB
  _Float16* big16   = (_Float16*)(ws + (size_t)67108864);          // 67 MB (qkv/ffh/rskv)
  char*     att     = ws + (size_t)134217728;                      // 33.5 MB region
  float*    attnout = (float*)att;                                 // encoder phase
  float*    qbc     = (float*)(att);                               // rs phase (attnout dead)
  float*    rsheads = (float*)(att + (1u << 20));
  float*    memA    = (float*)(att + (2u << 20));
  float*    memB    = (float*)(att + (3u << 20));
  _Float16* rsq     = (_Float16*)(att + (4u << 20));
  _Float16* ffrs    = (_Float16*)(att + (5u << 20));
  _Float16* kg      = (_Float16*)(att + (8u << 20));
  _Float16* vg      = (_Float16*)(att + (12u << 20));
  _Float16* qkv     = big16;   // [BS, 3D] f16
  _Float16* ffh     = big16;   // [BS, 4D] f16 (after attention)
  _Float16* rskv    = big16;   // [BS, 2D] f16 (resampler)

  // ---- embedding gather fused into input projection GEMM ----
  wmma_gemm_kernel<float, false, float><<<dim3(C_D / 64, C_BS / 64), 128, 0, stream>>>(
      emb, ids, ipw, ipb, nullptr, x, C_BS, C_D, C_E);

  // ---- pre-norm transformer encoder ----
  for (int l = 0; l < C_LENC; l++) {
    layernorm_kernel<<<C_BS, 128, 0, stream>>>(x, ln1g + l * C_D, ln1b + l * C_D, h);
    wmma_gemm_kernel<float, false, _Float16><<<dim3(3 * C_D / 64, C_BS / 64), 128, 0, stream>>>(
        h, nullptr, enc_in_w + (size_t)l * 3 * C_D * C_D, enc_in_b + l * 3 * C_D,
        nullptr, qkv, C_BS, 3 * C_D, C_D);
    flash_attn_kernel<<<dim3(C_S / 16, C_B * C_NH), 32, 0, stream>>>(
        qkv, 3 * C_D, qkv + C_D, qkv + 2 * C_D, 3 * C_D,
        attnout, C_D, C_S, C_S, C_NH, 0.125f);
    wmma_gemm_kernel<float, false, float><<<dim3(C_D / 64, C_BS / 64), 128, 0, stream>>>(
        attnout, nullptr, enc_ow + (size_t)l * C_D * C_D, enc_ob + l * C_D,
        x, x, C_BS, C_D, C_D);
    layernorm_kernel<<<C_BS, 128, 0, stream>>>(x, ln2g + l * C_D, ln2b + l * C_D, h);
    wmma_gemm_kernel<float, true, _Float16><<<dim3(4 * C_D / 64, C_BS / 64), 128, 0, stream>>>(
        h, nullptr, ff1w + (size_t)l * 4 * C_D * C_D, ff1b + l * 4 * C_D,
        nullptr, ffh, C_BS, 4 * C_D, C_D);
    wmma_gemm_kernel<_Float16, false, float><<<dim3(C_D / 64, C_BS / 64), 128, 0, stream>>>(
        ffh, nullptr, ff2w + (size_t)l * C_D * 4 * C_D, ff2b + l * C_D,
        x, x, C_BS, C_D, 4 * C_D);
  }

  // ---- Perceiver resampler (post-norm) ----
  const int BM = C_B * C_M;  // 256
  bcast_queries_kernel<<<BM, 128, 0, stream>>>(queries, qbc);
  wmma_gemm_kernel<float, false, _Float16><<<dim3(C_D / 64, BM / 64), 128, 0, stream>>>(
      qbc, nullptr, rs_in_w, rs_in_b, nullptr, rsq, BM, C_D, C_D);
  wmma_gemm_kernel<float, false, _Float16><<<dim3(2 * C_D / 64, C_BS / 64), 128, 0, stream>>>(
      x, nullptr, rs_in_w + (size_t)C_D * C_D, rs_in_b + C_D,
      nullptr, rskv, C_BS, 2 * C_D, C_D);
  flash_attn_kernel<<<dim3(1, C_B * C_NH), 32, 0, stream>>>(
      rsq, C_D, rskv, rskv + C_D, 2 * C_D,
      rsheads, C_D, C_M, C_S, C_NH, 0.125f);
  wmma_gemm_kernel<float, false, float><<<dim3(C_D / 64, BM / 64), 128, 0, stream>>>(
      rsheads, nullptr, rs_ow, rs_ob, qbc, memA, BM, C_D, C_D);
  layernorm_kernel<<<BM, 128, 0, stream>>>(memA, rl1g, rl1b, memB);
  wmma_gemm_kernel<float, true, _Float16><<<dim3(4 * C_D / 64, BM / 64), 128, 0, stream>>>(
      memB, nullptr, rf1w, rf1b, nullptr, ffrs, BM, 4 * C_D, C_D);
  wmma_gemm_kernel<_Float16, false, float><<<dim3(C_D / 64, BM / 64), 128, 0, stream>>>(
      ffrs, nullptr, rf2w, rf2b, memB, memA, BM, C_D, 4 * C_D);
  layernorm_kernel<<<BM, 128, 0, stream>>>(memA, rl2g, rl2b, memB);

  // ---- per-group K/V projections ----
  for (int g = 0; g < C_G; g++) {
    wmma_gemm_kernel<float, false, _Float16><<<dim3(C_KV / 64, BM / 64), 128, 0, stream>>>(
        memB, nullptr, pkw + (size_t)g * C_KV * C_D, pkb + g * C_KV,
        nullptr, kg + (size_t)g * BM * C_KV, BM, C_KV, C_D);
    wmma_gemm_kernel<float, false, _Float16><<<dim3(C_KV / 64, BM / 64), 128, 0, stream>>>(
        memB, nullptr, pvw + (size_t)g * C_KV * C_D, pvb + g * C_KV,
        nullptr, vg + (size_t)g * BM * C_KV, BM, C_KV, C_D);
  }

  // ---- gather per layer, gate, cast fp16 into d_out (2,LL,B,KVH,M,HD) ----
  const int total_chunks = 2 * C_LL * C_B * C_KVH * C_M * (C_HD / 8);  // 3145728
  gate_out_kernel<<<(total_chunks + 255) / 256, 256, 0, stream>>>(
      kg, vg, gates, l2g, (_Float16*)d_out, total_chunks);
}
